// Axis_Portrait_Attention_block_9706626089299
// MI455X (gfx1250) — matrix operand order repaired and checked by reading, not compiled, not run on hardware
//
#include <hip/hip_runtime.h>

// Problem constants (B,C,H,W) = (4,256,128,128), RATION=1 -> Cr=256
#define B_   4
#define C_   256
#define H_   128
#define W_   128
#define HW_  (H_ * W_)       // 16384
#define CHW_ (C_ * HW_)      // 4194304 elements per batch
#define CH_  (C_ * H_)       // 32768 — attention contraction length

typedef __attribute__((ext_vector_type(16))) __bf16 v16bf;
typedef __attribute__((ext_vector_type(8)))  float  v8f;

// ---- gfx1250 async global->LDS path (guarded; falls back to manual copy) ----
#if defined(__gfx1250__) && \
    __has_builtin(__builtin_amdgcn_global_load_async_to_lds_b128) && \
    __has_builtin(__builtin_amdgcn_s_wait_asynccnt)
#define HAVE_ASYNC_LDS 1
// The builtin's params are (global int4-vec*, shared int4-vec*, imm off, imm cpol)
typedef int v4i_gcc __attribute__((vector_size(4 * sizeof(int))));
typedef __attribute__((address_space(1))) v4i_gcc* gv4i_p;
typedef __attribute__((address_space(3))) v4i_gcc* lv4i_p;
#else
#define HAVE_ASYNC_LDS 0
#endif

// Copy 16 bf16 (32B) global -> LDS. Both pointers 16B-aligned.
__device__ inline void copy16_g2l(__bf16* dst, const __bf16* src) {
#if HAVE_ASYNC_LDS
  __builtin_amdgcn_global_load_async_to_lds_b128((gv4i_p)src, (lv4i_p)dst, 0, 0);
  __builtin_amdgcn_global_load_async_to_lds_b128((gv4i_p)(src + 8),
                                                 (lv4i_p)(dst + 8), 0, 0);
#else
#pragma unroll
  for (int e = 0; e < 16; ++e) dst[e] = src[e];
#endif
}
__device__ inline void async_copy_fence() {
#if HAVE_ASYNC_LDS
  __builtin_amdgcn_s_wait_asynccnt(0);   // LDS writes landed before barrier
#endif
}

// ---- WMMA fragment builders -------------------------------------------------
// A-matrix 16x32 bf16 layout: lane L (m = L&15, hl = L>>4):
//   elements 0..7  <- K = hl*8 + e        (VGPR0..3)
//   elements 8..15 <- K = 16 + hl*8 + e   (VGPR4..7)
__device__ inline v16bf frag_a_f32(const float* __restrict__ row, int hl) {
  v16bf a;
#pragma unroll
  for (int e = 0; e < 8; ++e) {
    a[e]     = (__bf16)row[hl * 8 + e];
    a[e + 8] = (__bf16)row[16 + hl * 8 + e];
  }
  return a;
}
__device__ inline v16bf frag_a_bf16(const __bf16* __restrict__ row, int hl) {
  v16bf a;
#pragma unroll
  for (int e = 0; e < 8; ++e) {
    a[e]     = row[hl * 8 + e];
    a[e + 8] = row[16 + hl * 8 + e];
  }
  return a;
}
// B-matrix 32x16 bf16 layout: lane L (n = L&15, hl = L>>4):
//   elements 0..7  <- K = hl*8 + e; elements 8..15 <- K = 16 + hl*8 + e
__device__ inline v16bf frag_b_bf16(const __bf16* __restrict__ col, int hl) {
  v16bf b;
#pragma unroll
  for (int e = 0; e < 8; ++e) {
    b[e]     = col[hl * 8 + e];
    b[e + 8] = col[16 + hl * 8 + e];
  }
  return b;
}
__device__ inline v8f wmma_bf16(v16bf a, v16bf b, v8f c) {
  return __builtin_amdgcn_wmma_f32_16x16x32_bf16(false, a, false, b, (short)0, c,
                                                 false, false);
}

// ---- Kernel 1: Q/K/V = relu(W x + b), GEMM [256 x 256] x [256 x 16384] ------
// grid: (s-band 128, o-band 2, mat*B 12). Writes bf16 ws + fp32 outputs
// (x_Qw direct; x_Kw / x_Vw per-batch transposed).
__global__ void __launch_bounds__(256) k_qkv(
    const float* __restrict__ x,
    const float* __restrict__ Wq, const float* __restrict__ bq,
    const float* __restrict__ Wk, const float* __restrict__ bk,
    const float* __restrict__ Wv, const float* __restrict__ bv,
    __bf16* __restrict__ ws_qkv,   // [3][B][C][HW] bf16
    float*  __restrict__ out_qkv)  // x_Qw | x_Kw | x_Vw fp32 regions
{
  __shared__ __bf16 Xt[128][32];   // x tile transposed: [s][c]
  const int mat   = blockIdx.z % 3;
  const int b     = blockIdx.z / 3;
  const int oBase = blockIdx.y * 128;
  const int sBase = blockIdx.x * 128;

  const float* Wm = (mat == 0) ? Wq : (mat == 1) ? Wk : Wv;
  const float* bm = (mat == 0) ? bq : (mat == 1) ? bk : bv;
  __bf16* dstb = ws_qkv + ((size_t)mat * B_ + b) * CHW_;
  float*  dstf = out_qkv + ((size_t)mat * B_ + b) * CHW_;

  const int tid  = threadIdx.x;
  const int lane = tid & 31, wave = tid >> 5;
  const int l15 = lane & 15, hl = lane >> 4;
  const int oSub = (wave & 3) * 32;   // 4 waves along o
  const int sSub = (wave >> 2) * 64;  // 2 waves along s

  v8f acc[2][4] = {};

  for (int kc = 0; kc < C_; kc += 32) {
    { // stage x[c, s] -> Xt[s][c] in bf16 (transpose for B fragments)
      const int c    = tid >> 3;          // 0..31
      const int sOff = (tid & 7) * 16;    // 0..112
      const float* src = x + (size_t)(b * C_ + kc + c) * HW_ + sBase + sOff;
#pragma unroll
      for (int e = 0; e < 16; ++e) Xt[sOff + e][c] = (__bf16)src[e];
    }
    __syncthreads();

    v16bf afr[2];
#pragma unroll
    for (int tm = 0; tm < 2; ++tm) {
      const int orow = oBase + oSub + tm * 16 + l15;
      afr[tm] = frag_a_f32(Wm + (size_t)orow * C_ + kc, hl);
    }
#pragma unroll
    for (int tn = 0; tn < 4; ++tn) {
      const int scol = sSub + tn * 16 + l15;
      v16bf bfr = frag_b_bf16(&Xt[scol][0], hl);
#pragma unroll
      for (int tm = 0; tm < 2; ++tm)
        acc[tm][tn] = wmma_bf16(afr[tm], bfr, acc[tm][tn]);
    }
    __syncthreads();
  }

  // epilogue: bias + relu; C/D layout: m = r + hl*8, n = l15
#pragma unroll
  for (int tm = 0; tm < 2; ++tm)
#pragma unroll
    for (int tn = 0; tn < 4; ++tn)
#pragma unroll
      for (int r = 0; r < 8; ++r) {
        const int o = oBase + oSub + tm * 16 + r + hl * 8;
        const int s = sBase + sSub + tn * 16 + l15;
        float v = fmaxf(acc[tm][tn][r] + bm[o], 0.0f);
        const int flat = (o << 14) + s;        // o*HW + s
        dstb[flat] = (__bf16)v;
        if (mat == 0) {
          dstf[flat] = v;                      // x_Qw == raw Q layout
        } else {                               // x_Kw/x_Vw = per-batch transpose
          const int j  = flat >> 15;           // row of the [128,32768] reshape
          const int kk = flat & (CH_ - 1);
          dstf[(size_t)kk * W_ + j] = v;
        }
      }
}

// ---- Kernel 2: S = Qr @ Kr^T, softmax over axis=1 (columns) -----------------
// One block per batch; wave w owns the 16-column band j = 16w for ALL 128 rows,
// so the column softmax reduces in-wave (lane L pairs with L^16 via shfl_xor).
__global__ void __launch_bounds__(256) k_attn(
    const __bf16* __restrict__ Qb,   // [B][128][32768] bf16
    const __bf16* __restrict__ Kb,
    float*  __restrict__ attn_out,   // [B][128][128] fp32
    __bf16* __restrict__ attnT)      // [B][j=128][i=128] bf16 (for next GEMM)
{
  __shared__ __bf16 Qs[128][32];
  const int b = blockIdx.x;
  const int tid = threadIdx.x;
  const int lane = tid & 31, wave = tid >> 5;
  const int l15 = lane & 15, hl = lane >> 4;
  const int jband = wave * 16;

  const __bf16* Qbb = Qb + (size_t)b * CHW_;
  const __bf16* Kbb = Kb + (size_t)b * CHW_;

  v8f acc[8] = {};

  for (int kk = 0; kk < CH_; kk += 32) {
    { // stage Q tile [i=128][k=32] via async global->LDS (bf16 verbatim)
      const int i  = tid >> 1;
      const int hf = (tid & 1) * 16;
      const __bf16* src = Qbb + (size_t)i * CH_ + kk + hf;
      __builtin_prefetch(src + 32, 0, 1);      // global_prefetch_b8 next tile
      copy16_g2l(&Qs[i][hf], src);
      async_copy_fence();
    }
    __syncthreads();

    const int j = jband + l15;                 // K rows are K-contiguous -> B frag
    v16bf bfr = frag_b_bf16(Kbb + (size_t)j * CH_ + kk, hl);
#pragma unroll
    for (int f = 0; f < 8; ++f) {
      v16bf afr = frag_a_bf16(&Qs[f * 16 + l15][0], hl);
      acc[f] = wmma_bf16(afr, bfr, acc[f]);
    }
    __syncthreads();
  }

  // softmax over i (column-wise): lane holds i = f*16 + r + hl*8 for fixed j
  float mx = -3.4e38f;
#pragma unroll
  for (int f = 0; f < 8; ++f)
#pragma unroll
    for (int r = 0; r < 8; ++r) mx = fmaxf(mx, acc[f][r]);
  mx = fmaxf(mx, __shfl_xor(mx, 16, 32));

  float sum = 0.0f;
#pragma unroll
  for (int f = 0; f < 8; ++f)
#pragma unroll
    for (int r = 0; r < 8; ++r) {
      float e = __expf(acc[f][r] - mx);
      acc[f][r] = e;
      sum += e;
    }
  sum += __shfl_xor(sum, 16, 32);
  const float inv = 1.0f / sum;

  const int j = jband + l15;
  float*  aout = attn_out + (size_t)b * (W_ * W_);
  __bf16* atT  = attnT    + (size_t)b * (W_ * W_);
#pragma unroll
  for (int f = 0; f < 8; ++f)
#pragma unroll
    for (int r = 0; r < 8; ++r) {
      const int i = f * 16 + r + hl * 8;
      const float p = acc[f][r] * inv;
      aout[i * W_ + j] = p;
      atT[j * W_ + i]  = (__bf16)p;
    }
}

// ---- Kernel 3: out1^T = attn^T @ Vr  ---------------------------------------
// grid: (kband 256, batch 4). V staged transposed in LDS so B frags get
// i-contiguous runs. Output scattered to [b][h][w][c] so kernel 4's B operand
// is c-contiguous.
__global__ void __launch_bounds__(256) k_av(
    const __bf16* __restrict__ Vb,     // [B][128 i][32768 k]
    const __bf16* __restrict__ attnT,  // [B][128 j][128 i]
    __bf16* __restrict__ out1c)        // [B][128 h][128 w][256 c]
{
  __shared__ __bf16 Vt[128][128];      // [k_local][i]
  const int b = blockIdx.y;
  const int kbase = blockIdx.x * 128;
  const int tid = threadIdx.x;
  const int lane = tid & 31, wave = tid >> 5;
  const int l15 = lane & 15, hl = lane >> 4;

  const __bf16* Vbb = Vb + (size_t)b * CHW_;
  const __bf16* atT = attnT + (size_t)b * (W_ * W_);

  { // stage V tile transposed (element shuffle -> manual path)
    const int i  = tid >> 1;
    const int hf = (tid & 1) * 64;
    const __bf16* src = Vbb + (size_t)i * CH_ + kbase + hf;
#pragma unroll
    for (int e = 0; e < 64; ++e) Vt[hf + e][i] = src[e];
  }
  __syncthreads();

  v8f acc[8] = {};
  const int jrow = wave * 16 + l15;    // A rows (j) — i-contiguous in attnT
  for (int i0 = 0; i0 < 128; i0 += 32) {
    v16bf afr = frag_a_bf16(atT + (size_t)jrow * 128 + i0, hl);
#pragma unroll
    for (int kt = 0; kt < 8; ++kt) {
      v16bf bfr = frag_b_bf16(&Vt[kt * 16 + l15][i0], hl);
      acc[kt] = wmma_bf16(afr, bfr, acc[kt]);
    }
  }

  // D: m -> j (= w), n -> k; k = c*128 + h
#pragma unroll
  for (int kt = 0; kt < 8; ++kt)
#pragma unroll
    for (int r = 0; r < 8; ++r) {
      const int w = wave * 16 + r + hl * 8;
      const int k = kbase + kt * 16 + l15;
      const int c = k >> 7;
      const int h = k & 127;
      out1c[(size_t)b * CHW_ + ((size_t)h * 128 + w) * 256 + c] = (__bf16)acc[kt][r];
    }
}

// ---- Kernel 4: out_w = Wu @ (gamma * out1) + bu -----------------------------
// grid: (h 128, o-band 2, batch 4). B tile [w][c] fully staged (c-contiguous)
// via async global->LDS.
__global__ void __launch_bounds__(256) k_proj(
    const __bf16* __restrict__ out1c,  // [B][h][w][c]
    const float* __restrict__ Wu, const float* __restrict__ bu,
    const float* __restrict__ gamma,
    float* __restrict__ out_w,         // [B][C][H][W]
    float* __restrict__ gamma_out)
{
  extern __shared__ __bf16 Bw[];       // [128 w][256 c] = 64 KB
  const int b = blockIdx.z;
  const int oBase = blockIdx.y * 128;
  const int h = blockIdx.x;
  const int tid = threadIdx.x;
  const int lane = tid & 31, wave = tid >> 5;
  const int l15 = lane & 15, hl = lane >> 4;
  const int oSub = (wave & 3) * 32;
  const int wSub = (wave >> 2) * 64;

  if (b == 0 && oBase == 0 && h == 0 && tid == 0) gamma_out[0] = gamma[0];

  { // stage B tile: contiguous 128-element rows, async copy
    const int w  = tid >> 1;
    const int hf = (tid & 1) * 128;
    const __bf16* src = out1c + (size_t)b * CHW_ + ((size_t)h * 128 + w) * 256 + hf;
#pragma unroll
    for (int e = 0; e < 128; e += 16) copy16_g2l(&Bw[w * 256 + hf + e], src + e);
    async_copy_fence();
  }
  __syncthreads();

  const float g = gamma[0];
  v8f acc[2][4] = {};
  for (int c0 = 0; c0 < C_; c0 += 32) {
    v16bf afr[2];
#pragma unroll
    for (int tm = 0; tm < 2; ++tm) {
      const int orow = oBase + oSub + tm * 16 + l15;
      afr[tm] = frag_a_f32(Wu + (size_t)orow * C_ + c0, hl);
    }
#pragma unroll
    for (int tn = 0; tn < 4; ++tn) {
      const int wcol = wSub + tn * 16 + l15;
      v16bf bfr = frag_b_bf16(&Bw[wcol * 256 + c0], hl);
#pragma unroll
      for (int tm = 0; tm < 2; ++tm)
        acc[tm][tn] = wmma_bf16(afr[tm], bfr, acc[tm][tn]);
    }
  }

#pragma unroll
  for (int tm = 0; tm < 2; ++tm)
#pragma unroll
    for (int tn = 0; tn < 4; ++tn)
#pragma unroll
      for (int r = 0; r < 8; ++r) {
        const int o = oBase + oSub + tm * 16 + r + hl * 8;
        const int w = wSub + tn * 16 + l15;
        out_w[(size_t)b * CHW_ + (size_t)o * HW_ + h * W_ + w] =
            bu[o] + g * acc[tm][tn][r];
      }
}

// ---- host-side launcher -----------------------------------------------------
extern "C" void kernel_launch(void* const* d_in, const int* in_sizes, int n_in,
                              void* d_out, int out_size, void* d_ws, size_t ws_size,
                              hipStream_t stream) {
  const float* x  = (const float*)d_in[0];
  const float* Wq = (const float*)d_in[1];
  const float* bq = (const float*)d_in[2];
  const float* Wk = (const float*)d_in[3];
  const float* bk = (const float*)d_in[4];
  const float* Wv = (const float*)d_in[5];
  const float* bv = (const float*)d_in[6];
  const float* Wu = (const float*)d_in[7];
  const float* bu = (const float*)d_in[8];
  const float* gm = (const float*)d_in[9];

  // Output regions (fp32), concatenated in return order
  float* out    = (float*)d_out;
  float* out_w  = out;                                 // [4,256,128,128]
  float* out_q  = out + (size_t)B_ * CHW_;             // x_Qw | x_Kw | x_Vw (contiguous)
  float* out_g  = out + (size_t)4 * B_ * CHW_;         // gamma (1)
  float* out_at = out_g + 1;                           // attn [4,128,128]

  // Workspace (bf16): Q | K | V | out1c | attnT  ~= 128 MB
  __bf16* ws    = (__bf16*)d_ws;
  __bf16* qkvb  = ws;                                  // 3 * B*CHW
  __bf16* out1c = ws + (size_t)3 * B_ * CHW_;          // B*CHW
  __bf16* attnT = ws + (size_t)4 * B_ * CHW_;          // B*128*128

  dim3 blk(256);
  k_qkv <<<dim3(128, 2, 12), blk, 0, stream>>>(x, Wq, bq, Wk, bk, Wv, bv,
                                               qkvb, out_q);
  k_attn<<<dim3(4, 1, 1),   blk, 0, stream>>>(qkvb,                         // Q
                                              qkvb + (size_t)B_ * CHW_,     // K
                                              out_at, attnT);
  k_av  <<<dim3(256, 4, 1), blk, 0, stream>>>(qkvb + (size_t)2 * B_ * CHW_, // V
                                              attnT, out1c);
  k_proj<<<dim3(128, 2, 4), blk, 65536, stream>>>(out1c, Wu, bu, gm, out_w, out_g);
}
